// TopKSAE_87797721465032
// MI455X (gfx1250) — matrix operand order repaired and checked by reading, not compiled, not run on hardware
//
#include <hip/hip_runtime.h>
#include <hip/hip_bf16.h>

// ---------------------------------------------------------------------------
// TopK SAE forward for gfx1250 (MI455X).
// d_out layout: x_hat[8192*768] | latents[8192*6144] | loss[1] | aux_loss[1]
// ---------------------------------------------------------------------------

#define D_MODEL 768
#define D_SAE   6144
#define BATCH   8192
#define TOPK    32

typedef __attribute__((ext_vector_type(16))) __bf16        v16bf;
typedef __attribute__((ext_vector_type(8)))  float         v8f;
typedef __attribute__((ext_vector_type(8)))  unsigned int  v8u;

__device__ __forceinline__ unsigned short f2bf(float f) {
    unsigned int u = __float_as_uint(f);
    // round-to-nearest-even truncation to bf16
    unsigned int r = (u + 0x7FFFu + ((u >> 16) & 1u)) >> 16;
    return (unsigned short)r;
}

// ---------------------------------------------------------------------------
// Kernel 1: pre_acts = x @ W_enc + b_enc  (written into latents region)
// Block: 256 threads (8 waves). Tile: 128 (M) x 128 (N). K staged 32 at a
// time through LDS as bf16; each wave computes a 16x128 strip via 8 WMMA
// tiles off a single A fragment per K-step.
// ---------------------------------------------------------------------------
__global__ void __launch_bounds__(256)
sae_enc_gemm(const float* __restrict__ x, const float* __restrict__ W,
             const float* __restrict__ b_enc, float* __restrict__ pre) {
    __shared__ unsigned short As[128][32];   // [m][k] bf16
    __shared__ unsigned short Bs[128][36];   // [n][k] bf16 (K-transposed, padded)

    const int tid  = threadIdx.x;
    const int wv   = tid >> 5;      // wave 0..7
    const int lane = tid & 31;
    const int hv   = lane >> 4;     // lane half (0/1)
    const int l16  = lane & 15;
    const int rowBase = blockIdx.y * 128;
    const int colBase = blockIdx.x * 128;

    v8f c[8];
    #pragma unroll
    for (int t = 0; t < 8; ++t) c[t] = (v8f)0.0f;

    for (int k0 = 0; k0 < D_MODEL; k0 += 32) {
        // ---- stage A: 128x32 floats -> bf16 LDS (1024 float4, 4/thread) ----
        #pragma unroll
        for (int i = 0; i < 4; ++i) {
            int f4 = tid + i * 256;               // 0..1023
            int r  = f4 >> 3;                     // row 0..127
            int cc = (f4 & 7) * 4;                // col 0..28
            float4 v = *(const float4*)(x + (size_t)(rowBase + r) * D_MODEL + k0 + cc);
            As[r][cc + 0] = f2bf(v.x);
            As[r][cc + 1] = f2bf(v.y);
            As[r][cc + 2] = f2bf(v.z);
            As[r][cc + 3] = f2bf(v.w);
        }
        // ---- stage B: 32x128 floats -> bf16 LDS transposed (1024 f4) -------
        #pragma unroll
        for (int i = 0; i < 4; ++i) {
            int f4 = tid + i * 256;               // 0..1023
            int r  = f4 >> 5;                     // k row 0..31
            int cc = (f4 & 31) * 4;               // n col 0..124
            float4 v = *(const float4*)(W + (size_t)(k0 + r) * D_SAE + colBase + cc);
            Bs[cc + 0][r] = f2bf(v.x);
            Bs[cc + 1][r] = f2bf(v.y);
            Bs[cc + 2][r] = f2bf(v.z);
            Bs[cc + 3][r] = f2bf(v.w);
        }
        __syncthreads();

        // ---- A fragment (16x32 bf16, ISA lane layout) ----
        int am = wv * 16 + l16;
        v8u au;
        #pragma unroll
        for (int v = 0; v < 8; ++v) {
            int kk = (v < 4 ? 2 * v : 16 + 2 * (v - 4)) + 8 * hv;  // even
            au[v] = *(const unsigned int*)&As[am][kk];
        }
        v16bf a = __builtin_bit_cast(v16bf, au);

        // ---- 8 B fragments (32x16 bf16 each) + WMMA ----
        #pragma unroll
        for (int t = 0; t < 8; ++t) {
            int bn = t * 16 + l16;
            v8u bu;
            #pragma unroll
            for (int v = 0; v < 8; ++v) {
                int kk = (v < 4 ? 2 * v : 16 + 2 * (v - 4)) + 8 * hv;  // even
                bu[v] = *(const unsigned int*)&Bs[bn][kk];
            }
            v16bf b = __builtin_bit_cast(v16bf, bu);
            c[t] = __builtin_amdgcn_wmma_f32_16x16x32_bf16(
                false, a, false, b, (short)0, c[t], false, false);
        }
        __syncthreads();
    }

    // ---- epilogue: D layout  m = r + 8*hv, n = l16 ----
    #pragma unroll
    for (int t = 0; t < 8; ++t) {
        int col = colBase + t * 16 + l16;
        float bias = b_enc[col];
        #pragma unroll
        for (int r = 0; r < 8; ++r) {
            int row = rowBase + wv * 16 + r + 8 * hv;
            pre[(size_t)row * D_SAE + col] = c[t][r] + bias;
        }
    }
}

// ---------------------------------------------------------------------------
// Kernel 2: per-row top-64 (32 latents + 32 aux) via candidate cache.
// Each thread owns a 24-element strided slice and posts its local max; each
// pick reduces the 256 cached candidates (wave 0) and only the owner thread
// rescans its slice. In-place sparsification of the latents region;
// (idx, relu(val)) pairs to workspace.
// ---------------------------------------------------------------------------
__global__ void __launch_bounds__(256)
sae_topk(float* __restrict__ latents, int* __restrict__ sel_idx,
         float* __restrict__ sel_val) {
    __shared__ float pre[D_SAE];
    __shared__ float cval[256];
    __shared__ int   cidx[256];
    __shared__ float s_val[64];
    __shared__ int   s_idx[64];
    __shared__ int   s_owner;

    const int row = blockIdx.x;
    const int tid = threadIdx.x;
    float* g = latents + (size_t)row * D_SAE;
    const float NEG = -3.402823466e38f;

    for (int i = tid; i < D_SAE; i += 256) pre[i] = g[i];
    __syncthreads();

    auto rescan = [&](float& bv, int& bi) {
        bv = NEG; bi = 0x7FFFFFFF;
        for (int i = tid; i < D_SAE; i += 256) {
            float v = pre[i];
            if (v > bv) { bv = v; bi = i; }   // ascending i keeps lowest index
        }
    };

    float bv; int bi;
    rescan(bv, bi);
    cval[tid] = bv; cidx[tid] = bi;
    __syncthreads();

    for (int it = 0; it < 64; ++it) {
        // ---- wave 0 reduces the 256 candidates (lowest index on ties) ----
        if (tid < 32) {
            float mv = NEG; int mi = 0x7FFFFFFF; int mo = 0;
            #pragma unroll
            for (int j = 0; j < 8; ++j) {
                int t2 = tid + j * 32;
                float v = cval[t2]; int ix = cidx[t2];
                if (v > mv || (v == mv && ix < mi)) { mv = v; mi = ix; mo = t2; }
            }
            #pragma unroll
            for (int m = 16; m > 0; m >>= 1) {
                float ov = __shfl_xor(mv, m, 32);
                int   oi = __shfl_xor(mi, m, 32);
                int   oo = __shfl_xor(mo, m, 32);
                if (ov > mv || (ov == mv && oi < mi)) { mv = ov; mi = oi; mo = oo; }
            }
            if (tid == 0) {
                s_val[it] = mv; s_idx[it] = mi; s_owner = mo;
                pre[mi] = NEG;                        // remove picked element
            }
        }
        __syncthreads();
        // ---- only the owner's candidate is stale: refresh it ----
        if (tid == s_owner) {
            rescan(bv, bi);
            cval[tid] = bv; cidx[tid] = bi;
        }
        __syncthreads();
        if (it == 31) {
            // mask = (latents > 0): non-positive top-32 picks stay eligible
            // for the aux top-k (matches reference masking semantics).
            if (tid < 32 && s_val[tid] <= 0.0f) pre[s_idx[tid]] = s_val[tid];
            __syncthreads();
            rescan(bv, bi);                           // full candidate rebuild
            cval[tid] = bv; cidx[tid] = bi;
            __syncthreads();
        }
    }

    // ---- sparsify the output row: zeros except relu(top-32) ----
    for (int i = tid; i < D_SAE; i += 256) g[i] = 0.0f;
    __syncthreads();
    if (tid < 32) {
        float v = s_val[tid];
        g[s_idx[tid]] = v > 0.0f ? v : 0.0f;
    }
    if (tid < 64) {
        float v = s_val[tid];
        sel_idx[(size_t)row * 64 + tid] = s_idx[tid];
        sel_val[(size_t)row * 64 + tid] = v > 0.0f ? v : 0.0f;
    }
}

// ---------------------------------------------------------------------------
// Kernel 3: inv_norm[f] = 1 / max(||W_dec[f,:]||, 1e-12). One wave per row.
// ---------------------------------------------------------------------------
__global__ void __launch_bounds__(256)
sae_dec_norm(const float* __restrict__ Wdec, float* __restrict__ inv_norm) {
    const int wv   = threadIdx.x >> 5;
    const int lane = threadIdx.x & 31;
    const int f    = blockIdx.x * 8 + wv;           // grid = 768 -> 6144 rows
    const float* wr = Wdec + (size_t)f * D_MODEL;
    float s = 0.0f;
    for (int i = lane; i < D_MODEL; i += 32) { float v = wr[i]; s += v * v; }
    #pragma unroll
    for (int m = 16; m > 0; m >>= 1) s += __shfl_xor(s, m, 32);
    if (lane == 0) {
        float n = sqrtf(s);
        n = n > 1e-12f ? n : 1e-12f;
        inv_norm[f] = 1.0f / n;
    }
}

// ---------------------------------------------------------------------------
// Kernel 4: zero the loss accumulators (re-run every launch: graph-safe).
// ---------------------------------------------------------------------------
__global__ void sae_zero(float* __restrict__ acc) {
    if (threadIdx.x < 2) acc[threadIdx.x] = 0.0f;
}

// ---------------------------------------------------------------------------
// Kernel 5: sparse decode + both losses. One block per batch row.
//   x_hat  = latents @ Wd + b_dec          (Wd = row-normalized W_dec)
//   loss   = mean((x_hat - x)^2)
//   aux    = AUX_COEF * mean((aux_recon - (x - x_hat))^2)
// ---------------------------------------------------------------------------
__global__ void __launch_bounds__(256)
sae_decode(const float* __restrict__ x, const float* __restrict__ Wdec,
           const float* __restrict__ b_dec, const float* __restrict__ inv_norm,
           const int* __restrict__ sel_idx, const float* __restrict__ sel_val,
           float* __restrict__ x_hat, float* __restrict__ loss_acc) {
    __shared__ int   sidx[64];
    __shared__ float sval[64];
    __shared__ float red[256];

    const int row = blockIdx.x;
    const int tid = threadIdx.x;

    if (tid < 64) {
        int f = sel_idx[(size_t)row * 64 + tid];
        sidx[tid] = f;
        sval[tid] = sel_val[(size_t)row * 64 + tid] * inv_norm[f];
    }
    __syncthreads();

    float lsum = 0.0f, asum = 0.0f;
    for (int j = tid; j < D_MODEL; j += 256) {
        float acc  = b_dec[j];
        float aacc = 0.0f;
        #pragma unroll 8
        for (int i = 0; i < 32; ++i)
            acc  += sval[i]      * Wdec[(size_t)sidx[i]      * D_MODEL + j];
        #pragma unroll 8
        for (int i = 32; i < 64; ++i)
            aacc += sval[i]      * Wdec[(size_t)sidx[i]      * D_MODEL + j];
        float xv = x[(size_t)row * D_MODEL + j];
        x_hat[(size_t)row * D_MODEL + j] = acc;
        float d = acc - xv;               // x_hat - x
        lsum += d * d;
        float e = aacc + d;               // aux_recon - (x - x_hat)
        asum += e * e;
    }

    red[tid] = lsum; __syncthreads();
    for (int s = 128; s > 0; s >>= 1) {
        if (tid < s) red[tid] += red[tid + s];
        __syncthreads();
    }
    if (tid == 0)
        atomicAdd(&loss_acc[0], red[0] * (1.0f / ((float)BATCH * (float)D_MODEL)));
    __syncthreads();

    red[tid] = asum; __syncthreads();
    for (int s = 128; s > 0; s >>= 1) {
        if (tid < s) red[tid] += red[tid + s];
        __syncthreads();
    }
    if (tid == 0)
        atomicAdd(&loss_acc[1],
                  red[0] * ((1.0f / 32.0f) / ((float)BATCH * (float)D_MODEL)));
}

// ---------------------------------------------------------------------------
extern "C" void kernel_launch(void* const* d_in, const int* in_sizes, int n_in,
                              void* d_out, int out_size, void* d_ws, size_t ws_size,
                              hipStream_t stream) {
    const float* x     = (const float*)d_in[0];
    const float* W_enc = (const float*)d_in[1];
    const float* b_enc = (const float*)d_in[2];
    const float* W_dec = (const float*)d_in[3];
    const float* b_dec = (const float*)d_in[4];

    float* x_hat   = (float*)d_out;
    float* latents = x_hat + (size_t)BATCH * D_MODEL;
    float* losses  = latents + (size_t)BATCH * D_SAE;   // [loss, aux_loss]

    int*   sel_idx  = (int*)d_ws;
    float* sel_val  = (float*)(sel_idx + (size_t)BATCH * 64);
    float* inv_norm = sel_val + (size_t)BATCH * 64;

    // 1) encoder GEMM (bf16 WMMA, 128x128 tile) -> pre_acts into latents
    sae_enc_gemm<<<dim3(D_SAE / 128, BATCH / 128), 256, 0, stream>>>(
        x, W_enc, b_enc, latents);

    // 2) top-64 select + sparsify latents
    sae_topk<<<BATCH, 256, 0, stream>>>(latents, sel_idx, sel_val);

    // 3) decoder row norms
    sae_dec_norm<<<D_SAE / 8, 256, 0, stream>>>(W_dec, inv_norm);

    // 4) zero loss accumulators
    sae_zero<<<1, 32, 0, stream>>>(losses);

    // 5) sparse decode + losses
    sae_decode<<<BATCH, 256, 0, stream>>>(x, W_dec, b_dec, inv_norm,
                                          sel_idx, sel_val, x_hat, losses);
}
